// PointTransformerLayer_39857296507162
// MI455X (gfx1250) — matrix operand order repaired and checked by reading, not compiled, not run on hardware
//
#include <hip/hip_runtime.h>
#include <hip/hip_bf16.h>

typedef __attribute__((ext_vector_type(16))) _Float16 v16h;
typedef __attribute__((ext_vector_type(8)))  _Float16 v8h;
typedef __attribute__((ext_vector_type(8)))  float    v8f;

#define B_    4
#define N_    4096
#define KNN_  16
#define C_    64
#define EPS_  1e-5f

__device__ __forceinline__ v8f wmma_f16(v16h a, v16h b, v8f c) {
  // D = A(16x32 f16) * B(32x16 f16) + C(16x16 f32)
  return __builtin_amdgcn_wmma_f32_16x16x32_f16(false, a, false, b, (short)0, c,
                                                false, false);
}

__device__ __forceinline__ v16h cat16(v8h lo, v8h hi) {
  v16h r;
#pragma unroll
  for (int e = 0; e < 8; ++e) { r[e] = lo[e]; r[e + 8] = hi[e]; }
  return r;
}

__device__ __forceinline__ unsigned lds_off(const void* p) {
  // generic pointers to LDS carry the LDS byte offset in addr[31:0]
  return (unsigned)(uintptr_t)p;
}

// ---------------------------------------------------------------------------
// Kernel 1: brute-force kNN (top-16 smallest squared distances, self included)
// Point tiles are staged global->LDS with the gfx1250 async-to-LDS path
// (ASYNCcnt-tracked, no VGPR round trip). Also copies `points` to output #0.
// ---------------------------------------------------------------------------
__global__ __launch_bounds__(256) void knn_kernel(const float* __restrict__ pts,
                                                  int* __restrict__ idx,
                                                  float* __restrict__ out_pts) {
  __shared__ float sx[256], sy[256], sz[256];
  const int t = threadIdx.x;
  const int b = blockIdx.x / (N_ / 256);
  const int n = (blockIdx.x % (N_ / 256)) * 256 + t;
  const float* pb = pts + (size_t)b * N_ * 3;

  const float px = pb[n * 3 + 0];
  const float py = pb[n * 3 + 1];
  const float pz = pb[n * 3 + 2];

  // pass-through output #0
  out_pts[(size_t)b * N_ * 3 + n * 3 + 0] = px;
  out_pts[(size_t)b * N_ * 3 + n * 3 + 1] = py;
  out_pts[(size_t)b * N_ * 3 + n * 3 + 2] = pz;

  float bd[KNN_];
  int   bi[KNN_];
#pragma unroll
  for (int s = 0; s < KNN_; ++s) { bd[s] = 3.4e38f; bi[s] = 0; }

  const unsigned lx = lds_off(&sx[t]);
  const unsigned ly = lds_off(&sy[t]);
  const unsigned lz = lds_off(&sz[t]);

  for (int m0 = 0; m0 < N_; m0 += 256) {
    __syncthreads();   // previous tile fully consumed
    const unsigned long long ga =
        (unsigned long long)(uintptr_t)(pb + (size_t)(m0 + t) * 3);
    asm volatile(
        "global_load_async_to_lds_b32 %0, %3, off\n\t"
        "global_load_async_to_lds_b32 %1, %3, off offset:4\n\t"
        "global_load_async_to_lds_b32 %2, %3, off offset:8"
        :
        : "v"(lx), "v"(ly), "v"(lz), "v"(ga)
        : "memory");
    asm volatile("s_wait_asynccnt 0x0" ::: "memory");
    __syncthreads();
    for (int j = 0; j < 256; ++j) {
      const float dx = sx[j] - px, dy = sy[j] - py, dz = sz[j] - pz;
      float d = dx * dx + dy * dy + dz * dz;
      if (d < bd[KNN_ - 1]) {   // rare: insert into sorted register list
        int id = m0 + j;
#pragma unroll
        for (int s = 0; s < KNN_; ++s) {
          const bool sw = d < bd[s];
          const float nd = sw ? d : bd[s];
          const int   ni = sw ? id : bi[s];
          const float od = sw ? bd[s] : d;
          const int   oi = sw ? bi[s] : id;
          bd[s] = nd; bi[s] = ni; d = od; id = oi;
        }
      }
    }
  }
#pragma unroll
  for (int s = 0; s < KNN_; ++s)
    idx[((size_t)b * N_ + n) * KNN_ + s] = bi[s];
}

// ---------------------------------------------------------------------------
// Kernel 2: q/k/v = W(64x64) @ features(64xN) + bias, via WMMA f16->f32.
// One wave per 16-point tile; 4 M-tiles x 2 K-steps x 3 weight sets = 24 WMMAs.
// ---------------------------------------------------------------------------
__global__ __launch_bounds__(32) void qkv_kernel(const float* __restrict__ feat,
                                                 const float* __restrict__ Wq, const float* __restrict__ bq,
                                                 const float* __restrict__ Wk, const float* __restrict__ bk,
                                                 const float* __restrict__ Wv, const float* __restrict__ bv,
                                                 float* __restrict__ q, float* __restrict__ k, float* __restrict__ v) {
  const int lane = threadIdx.x;
  const int tile = blockIdx.x;               // 0 .. B*N/16-1
  const int b    = tile / (N_ / 16);
  const int n0   = (tile % (N_ / 16)) * 16;
  const int col  = lane & 15;
  const int half = lane >> 4;

  const float* fb = feat + (size_t)b * C_ * N_;

  // B operand: 32(in-ch) x 16(points); per lane: col = point, K = e + 16*half
  v16h bm0, bm1;
#pragma unroll
  for (int e = 0; e < 16; ++e) {
    bm0[e] = (_Float16)fb[(e + 16 * half) * N_ + n0 + col];
    bm1[e] = (_Float16)fb[(e + 16 * half + 32) * N_ + n0 + col];
  }

  const float* Ws[3]   = {Wq, Wk, Wv};
  const float* bs[3]   = {bq, bk, bv};
  float*       outs[3] = {q, k, v};

#pragma unroll
  for (int s = 0; s < 3; ++s) {
#pragma unroll
    for (int mt = 0; mt < 4; ++mt) {
      const int row = mt * 16 + col;   // A row M = lane%16 within tile
      v16h a0, a1;
#pragma unroll
      for (int e = 0; e < 16; ++e) {
        const int kA = e + 16 * half;   // A layout K index
        a0[e] = (_Float16)Ws[s][row * 64 + kA];
        a1[e] = (_Float16)Ws[s][row * 64 + kA + 32];
      }
      v8f acc = {};
      acc = wmma_f16(a0, bm0, acc);
      acc = wmma_f16(a1, bm1, acc);
#pragma unroll
      for (int r = 0; r < 8; ++r) {
        const int o = mt * 16 + r + 8 * half;            // D row M = r + 8*half
        outs[s][((size_t)b * C_ + o) * N_ + n0 + col] = acc[r] + bs[s][o];
      }
    }
  }
}

// ---------------------------------------------------------------------------
// Fused attention: one wave per point; the point's 16 neighbors are one WMMA
// N-tile. Weights are pre-converted to f16 in LDS once per block; activations
// are staged in LDS as f16 transposed [k][ch] so all WMMA operands are
// contiguous ds_load_b128s.
// ---------------------------------------------------------------------------
__device__ __forceinline__ void conv64_wmma(const _Float16 (*Wh)[72],
                                            const _Float16 (*ainT)[72],
                                            int lane, v8f acc[4]) {
  const int col  = lane & 15;
  const int half = lane >> 4;
  // B operand: b0[e] = ain[ch = e + 16*half][col] -> contiguous in ainT[col]
  const v16h b0 = cat16(*(const v8h*)&ainT[col][16 * half],
                        *(const v8h*)&ainT[col][16 * half + 8]);
  const v16h b1 = cat16(*(const v8h*)&ainT[col][32 + 16 * half],
                        *(const v8h*)&ainT[col][32 + 16 * half + 8]);
#pragma unroll
  for (int mt = 0; mt < 4; ++mt) {
    const int row = mt * 16 + col;
    // A layout: kA = e + 16*half (+32 for second K-step)
    const v16h a0 = cat16(*(const v8h*)&Wh[row][16 * half],
                          *(const v8h*)&Wh[row][16 * half + 8]);
    const v16h a1 = cat16(*(const v8h*)&Wh[row][32 + 16 * half],
                          *(const v8h*)&Wh[row][32 + 16 * half + 8]);
    acc[mt] = wmma_f16(a0, b0, acc[mt]);
    acc[mt] = wmma_f16(a1, b1, acc[mt]);
  }
}

__global__ __launch_bounds__(128) void attn_kernel(
    const float* __restrict__ points,
    const float* __restrict__ qb, const float* __restrict__ kb, const float* __restrict__ vb,
    const int* __restrict__ idx,
    const float* __restrict__ pe_w1, const float* __restrict__ pe_g1, const float* __restrict__ pe_b1,
    const float* __restrict__ pe_m1, const float* __restrict__ pe_v1,
    const float* __restrict__ pe_w2, const float* __restrict__ pe_bias2,
    const float* __restrict__ at_g1, const float* __restrict__ at_b1,
    const float* __restrict__ at_m1, const float* __restrict__ at_v1,
    const float* __restrict__ at_w1,
    const float* __restrict__ at_g2, const float* __restrict__ at_b2,
    const float* __restrict__ at_m2, const float* __restrict__ at_v2,
    const float* __restrict__ at_w2, const float* __restrict__ at_bias2,
    float* __restrict__ y) {
  __shared__ _Float16 s_w1h[64][72];      // at_w1 as f16 (row-major, padded)
  __shared__ _Float16 s_w2h[64][72];      // at_w2 as f16
  __shared__ _Float16 s_ainT[4][16][72];  // conv input, [kk][ch], f16, padded
  __shared__ float    s_nvr[4][64][17];   // n_v + r  (f32)
  __shared__ float    s_log[4][64][17];   // conv2 logits (f32)
  __shared__ float    s_r1[4][16][3];     // PE hidden (post BN/ReLU)
  __shared__ int      s_idx[4][KNN_];
  __shared__ float s_bn1s[64], s_bn1b[64], s_bn2s[64], s_bn2b[64];
  __shared__ float s_pes[3], s_peb[3];

  const int t    = threadIdx.x;
  const int w    = t >> 5;    // wave (point) within block
  const int lane = t & 31;

  __builtin_prefetch(at_w1, 0, 0);     // global_prefetch_b8
  __builtin_prefetch(at_w2, 0, 0);

  // once per block: weights -> f16 LDS, BN scale/shift tables
  for (int i = t; i < 64 * 64; i += 128) {
    const int r = i >> 6, c = i & 63;
    s_w1h[r][c] = (_Float16)at_w1[i];
    s_w2h[r][c] = (_Float16)at_w2[i];
  }
  if (t < 64) {
    const float s1 = at_g1[t] * rsqrtf(at_v1[t] + EPS_);
    s_bn1s[t] = s1; s_bn1b[t] = at_b1[t] - at_m1[t] * s1;
    const float s2 = at_g2[t] * rsqrtf(at_v2[t] + EPS_);
    s_bn2s[t] = s2; s_bn2b[t] = at_b2[t] - at_m2[t] * s2;
  }
  if (t < 3) {
    const float s = pe_g1[t] * rsqrtf(pe_v1[t] + EPS_);
    s_pes[t] = s; s_peb[t] = pe_b1[t] - pe_m1[t] * s;
  }

  const int point = blockIdx.x * 4 + w;
  const int b = point / N_;
  const int n = point % N_;

  if (lane < KNN_)
    s_idx[w][lane] = idx[((size_t)b * N_ + n) * KNN_ + lane];
  __syncthreads();

  const float cx = points[((size_t)b * N_ + n) * 3 + 0];
  const float cy = points[((size_t)b * N_ + n) * 3 + 1];
  const float cz = points[((size_t)b * N_ + n) * 3 + 2];

  // PE hidden layer: r1 = relu(bn(pe_w1 @ rel_xyz)), one neighbor per lane 0..15
  if (lane < KNN_) {
    const int j = s_idx[w][lane];
    const float rx = points[((size_t)b * N_ + j) * 3 + 0] - cx;
    const float ry = points[((size_t)b * N_ + j) * 3 + 1] - cy;
    const float rz = points[((size_t)b * N_ + j) * 3 + 2] - cz;
#pragma unroll
    for (int o = 0; o < 3; ++o) {
      float h = pe_w1[o * 3 + 0] * rx + pe_w1[o * 3 + 1] * ry + pe_w1[o * 3 + 2] * rz;
      h = fmaxf(h * s_pes[o] + s_peb[o], 0.f);
      s_r1[w][lane][o] = h;
    }
  }
  __syncthreads();

  // Elementwise front-end: r = pe_w2 @ r1 + bias2;  nvr = n_v + r;
  // a1 = relu(bn1(q - n_k + r))  -> staged f16, transposed [kk][ch].
  for (int i = lane; i < C_ * KNN_; i += 32) {
    const int ch = i >> 4;
    const int kk = i & 15;
    const int j  = s_idx[w][kk];
    const float r = pe_w2[ch * 3 + 0] * s_r1[w][kk][0] +
                    pe_w2[ch * 3 + 1] * s_r1[w][kk][1] +
                    pe_w2[ch * 3 + 2] * s_r1[w][kk][2] + pe_bias2[ch];
    const size_t base = ((size_t)b * C_ + ch) * N_;
    const float nk = kb[base + j];
    const float nv = vb[base + j];
    const float qv = qb[base + n];
    s_nvr[w][ch][kk] = nv + r;
    const float a0 = qv - nk + r;
    s_ainT[w][kk][ch] = (_Float16)fmaxf(a0 * s_bn1s[ch] + s_bn1b[ch], 0.f);
  }
  __syncthreads();

  const int col  = lane & 15;
  const int half = lane >> 4;

  // conv1: a2 = at_w1 @ a1   (8 WMMAs)
  v8f acc1[4] = {};
  conv64_wmma(s_w1h, s_ainT[w], lane, acc1);
  __syncthreads();
#pragma unroll
  for (int mt = 0; mt < 4; ++mt)
#pragma unroll
    for (int r = 0; r < 8; ++r) {
      const int ch = mt * 16 + r + 8 * half;
      s_ainT[w][col][ch] =
          (_Float16)fmaxf(acc1[mt][r] * s_bn2s[ch] + s_bn2b[ch], 0.f);
    }
  __syncthreads();

  // conv2: a4 = at_w2 @ a3 + at_bias2   (8 WMMAs) -> f32 logits
  v8f acc2[4] = {};
  conv64_wmma(s_w2h, s_ainT[w], lane, acc2);
#pragma unroll
  for (int mt = 0; mt < 4; ++mt)
#pragma unroll
    for (int r = 0; r < 8; ++r) {
      const int ch = mt * 16 + r + 8 * half;
      s_log[w][ch][col] = acc2[mt][r] + at_bias2[ch];
    }
  __syncthreads();

  // softmax over K per channel + weighted sum with nvr; 2 channels per lane
#pragma unroll
  for (int ci = 0; ci < 2; ++ci) {
    const int ch = lane * 2 + ci;
    float m = -3.4e38f;
#pragma unroll
    for (int kk = 0; kk < KNN_; ++kk) m = fmaxf(m, s_log[w][ch][kk]);
    float se = 0.f, out = 0.f;
#pragma unroll
    for (int kk = 0; kk < KNN_; ++kk) {
      const float e = __expf(s_log[w][ch][kk] - m);
      se  += e;
      out += e * s_nvr[w][ch][kk];
    }
    y[((size_t)b * C_ + ch) * N_ + n] = out / se;
  }
}

// ---------------------------------------------------------------------------
extern "C" void kernel_launch(void* const* d_in, const int* in_sizes, int n_in,
                              void* d_out, int out_size, void* d_ws, size_t ws_size,
                              hipStream_t stream) {
  const float* points   = (const float*)d_in[0];
  const float* features = (const float*)d_in[1];
  const float* Wq = (const float*)d_in[2];  const float* bq = (const float*)d_in[3];
  const float* Wk = (const float*)d_in[4];  const float* bk = (const float*)d_in[5];
  const float* Wv = (const float*)d_in[6];  const float* bv = (const float*)d_in[7];
  const float* pe_w1 = (const float*)d_in[8];
  const float* pe_g1 = (const float*)d_in[9];  const float* pe_b1 = (const float*)d_in[10];
  const float* pe_m1 = (const float*)d_in[11]; const float* pe_v1 = (const float*)d_in[12];
  const float* pe_w2 = (const float*)d_in[13]; const float* pe_bias2 = (const float*)d_in[14];
  const float* at_g1 = (const float*)d_in[15]; const float* at_b1 = (const float*)d_in[16];
  const float* at_m1 = (const float*)d_in[17]; const float* at_v1 = (const float*)d_in[18];
  const float* at_w1 = (const float*)d_in[19];
  const float* at_g2 = (const float*)d_in[20]; const float* at_b2 = (const float*)d_in[21];
  const float* at_m2 = (const float*)d_in[22]; const float* at_v2 = (const float*)d_in[23];
  const float* at_w2 = (const float*)d_in[24]; const float* at_bias2 = (const float*)d_in[25];

  const size_t qkv_elems = (size_t)B_ * C_ * N_;   // 1,048,576 floats each
  float* q = (float*)d_ws;
  float* k = q + qkv_elems;
  float* v = k + qkv_elems;
  int*   idx = (int*)(v + qkv_elems);              // B*N*16 ints

  float* out_pts = (float*)d_out;                  // (B,N,3)
  float* y = out_pts + (size_t)B_ * N_ * 3;        // (B,C,N)

  knn_kernel<<<B_ * (N_ / 256), 256, 0, stream>>>(points, idx, out_pts);
  qkv_kernel<<<B_ * (N_ / 16), 32, 0, stream>>>(features, Wq, bq, Wk, bk, Wv, bv, q, k, v);
  attn_kernel<<<(B_ * N_) / 4, 128, 0, stream>>>(
      points, q, k, v, idx,
      pe_w1, pe_g1, pe_b1, pe_m1, pe_v1, pe_w2, pe_bias2,
      at_g1, at_b1, at_m1, at_v1, at_w1,
      at_g2, at_b2, at_m2, at_v2, at_w2, at_bias2, y);
}
